// GRUwLinearModel_55387898249460
// MI455X (gfx1250) — compile-verified, matrix operand order checked by reading, not run on hardware
//
#include <hip/hip_runtime.h>
#include <hip/hip_bf16.h>

#define IN_SIZE 96
#define HIDDEN  128
#define G3      (3 * HIDDEN)   // 384
#define LIN_IN  32
#define T_STEPS 65536

typedef __attribute__((ext_vector_type(16))) _Float16 v16h;
typedef __attribute__((ext_vector_type(8)))  _Float16 v8h;
typedef __attribute__((ext_vector_type(8)))  float    v8f;

// Convert 16 f32s (two contiguous 8-float chunks, 16 floats apart) into the
// per-lane half-packing required by the 16-bit WMMA A/B VGPR layout:
// halves 0..7  = p[0..7]   (K(+0..7) for this lane's K-chunk)
// halves 8..15 = p[16..23] (K(+16..23))
__device__ inline v16h cvt_tile16(const float* __restrict__ p) {
    v16h t;
#pragma unroll
    for (int j = 0; j < 8; ++j) {
        t[j]     = (_Float16)p[j];
        t[j + 8] = (_Float16)p[16 + j];
    }
    return t;
}

// ---------------------------------------------------------------------------
// Phase 1: igates_all[T][384] = input_GRU[T][96] @ w_ih^T + b_ih
// One wave per 16x16 output tile, K=96 as 3 chained v_wmma_f32_16x16x32_f16.
// 4096 M-tiles * 24 N-tiles = 98304 waves = 12288 blocks of 8 waves.
// ---------------------------------------------------------------------------
__global__ __launch_bounds__(256)
void igates_gemm(const float* __restrict__ x,      // [T][96]
                 const float* __restrict__ w_ih,   // [384][96]
                 const float* __restrict__ b_ih,   // [384]
                 float* __restrict__ ig) {         // [T][384] (workspace)
    const int lane = threadIdx.x & 31;
    const int wid  = blockIdx.x * (blockDim.x >> 5) + (threadIdx.x >> 5);
    const int tm = wid / 24;                       // M tile (rows of T)
    const int tn = wid - tm * 24;                  // N tile (gate columns)

    const int mrow = tm * 16 + (lane & 15);        // A: M across lanes
    const int gcol = tn * 16 + (lane & 15);        // B: N across lanes
    const int kob  = (lane < 16) ? 0 : 8;          // hi half-wave owns K+8..15 / K+24..31

    v8f c = {};
#pragma unroll
    for (int kt = 0; kt < 3; ++kt) {
        const int koff = kt * 32 + kob;
        v16h a = cvt_tile16(x    + (size_t)mrow * IN_SIZE + koff);
        v16h b = cvt_tile16(w_ih + (size_t)gcol * IN_SIZE + koff);
        c = __builtin_amdgcn_wmma_f32_16x16x32_f16(
                false, a, false, b, (short)0, c, false, false);
    }

    const float bias  = b_ih[gcol];
    const int   rbase = tm * 16 + ((lane >> 4) << 3);  // D: VGPR v -> M = v + 8*(lane/16)
#pragma unroll
    for (int v = 0; v < 8; ++v)
        ig[(size_t)(rbase + v) * G3 + gcol] = c[v] + bias;
}

// ---------------------------------------------------------------------------
// Phase 2: sequential GRU scan, single workgroup (768 threads = 24 waves).
// w_hh lives permanently in VGPRs as f16 WMMA A-tiles (wave w owns rows
// 16w..16w+15, 4 K-tiles). Per step: build B by broadcasting h (all 16 cols
// equal h) from LDS, 4-deep WMMA accumulate, store D column 0 -> hgates in
// LDS, then threads 0..127 do the f32 gate math and wave 0 does the readout
// dot via shuffle reduction. igates/x_lin prefetched one step ahead.
// ---------------------------------------------------------------------------
__global__ __launch_bounds__(768)
void gru_scan(const float* __restrict__ ig,        // [T][384]
              const float* __restrict__ xlin,      // [T][32]
              const float* __restrict__ h0,        // [128]
              const float* __restrict__ w_hh,      // [384][128]
              const float* __restrict__ b_n,       // [128]
              const float* __restrict__ lin_bias,  // [1]
              float* __restrict__ out) {           // [T]
    __shared__ __align__(16) _Float16 h16s[HIDDEN];  // h in f16 (B operand feed)
    __shared__ float hgs[G3];                        // hgates = w_hh @ h

    const int tid  = threadIdx.x;
    const int lane = tid & 31;
    const int w    = tid >> 5;                     // wave id = M tile (0..23)
    const int kob  = (lane < 16) ? 0 : 8;

    // --- preload w_hh into f16 A-layout VGPRs (rows 16w + (lane&15)) ---
    const float* wr = w_hh + (size_t)(w * 16 + (lane & 15)) * HIDDEN;
    v16h a0 = cvt_tile16(wr + 0 * 32 + kob);
    v16h a1 = cvt_tile16(wr + 1 * 32 + kob);
    v16h a2 = cvt_tile16(wr + 2 * 32 + kob);
    v16h a3 = cvt_tile16(wr + 3 * 32 + kob);

    // --- per-thread recurrent state (thread t owns h[t] for t<128) ---
    float h_reg = 0.f, bnv = 0.f;
    if (tid < HIDDEN) {
        h_reg = h0[tid];
        bnv   = b_n[tid];
        h16s[tid] = (_Float16)h_reg;
    }
    const float lb = lin_bias[0];

    // prefetch step 0 inputs
    float igr = 0.f, igz = 0.f, ign = 0.f, xl = 0.f;
    if (tid < HIDDEN) {
        igr = ig[tid];
        igz = ig[HIDDEN + tid];
        ign = ig[2 * HIDDEN + tid];
    }
    if (tid < LIN_IN) xl = xlin[tid];
    __syncthreads();

    const bool dlane = ((lane & 15) == 0);         // lanes 0 and 16 hold column N=0
    const int  rbase = w * 16 + ((lane >> 4) << 3);

    for (int step = 0; step < T_STEPS; ++step) {
        // ---- hgates = w_hh @ h via WMMA (EXEC all-1s here) ----
        v8f c = {};
#pragma unroll
        for (int kc = 0; kc < 4; ++kc) {
            const v8h* p0 = (const v8h*)(h16s + kc * 32 + kob);
            const v8h* p1 = (const v8h*)(h16s + kc * 32 + kob + 16);
            v16h b = __builtin_shufflevector(*p0, *p1,
                      0, 1, 2, 3, 4, 5, 6, 7, 8, 9, 10, 11, 12, 13, 14, 15);
            v16h a = (kc == 0) ? a0 : (kc == 1) ? a1 : (kc == 2) ? a2 : a3;
            c = __builtin_amdgcn_wmma_f32_16x16x32_f16(
                    false, a, false, b, (short)0, c, false, false);
        }
        if (dlane) {
#pragma unroll
            for (int v = 0; v < 8; ++v) hgs[rbase + v] = c[v];
        }
        __syncthreads();

        // ---- prefetch step+1 inputs (overlaps next step's WMMA) ----
        float igr_n = 0.f, igz_n = 0.f, ign_n = 0.f, xl_n = 0.f;
        if (step + 1 < T_STEPS) {
            const float* igp = ig + (size_t)(step + 1) * G3;
            if (tid < HIDDEN) {
                igr_n = igp[tid];
                igz_n = igp[HIDDEN + tid];
                ign_n = igp[2 * HIDDEN + tid];
            }
            if (tid < LIN_IN) xl_n = xlin[(size_t)(step + 1) * LIN_IN + tid];
        }

        // ---- gate math (threads 0..127 = waves 0..3) ----
        if (tid < HIDDEN) {
            float r = 1.f / (1.f + __expf(-(igr + hgs[tid])));
            float z = 1.f / (1.f + __expf(-(igz + hgs[HIDDEN + tid])));
            float n = tanhf(ign + r * (hgs[2 * HIDDEN + tid] + bnv));
            h_reg = n + z * (h_reg - n);
            h16s[tid] = (_Float16)h_reg;
            // readout: out[t] = dot(h_new[:32], x_lin[t]) + bias (wave 0 only)
            if (tid < LIN_IN) {
                float p = h_reg * xl;
#pragma unroll
                for (int m = 16; m >= 1; m >>= 1) p += __shfl_xor(p, m, 32);
                if (tid == 0) out[step] = p + lb;
            }
        }
        __syncthreads();  // h16s ready for next step's B build

        igr = igr_n; igz = igz_n; ign = ign_n; xl = xl_n;
    }
}

// ---------------------------------------------------------------------------
extern "C" void kernel_launch(void* const* d_in, const int* in_sizes, int n_in,
                              void* d_out, int out_size, void* d_ws, size_t ws_size,
                              hipStream_t stream) {
    const float* x    = (const float*)d_in[0];  // input_GRU   [T][96]
    const float* xlin = (const float*)d_in[1];  // input_linear[T][32]
    const float* h0   = (const float*)d_in[2];  // init_hidden [128]
    const float* w_ih = (const float*)d_in[3];  // [384][96]
    const float* w_hh = (const float*)d_in[4];  // [384][128]
    const float* b_ih = (const float*)d_in[5];  // [384]
    const float* b_n  = (const float*)d_in[6];  // [128]
    const float* lb   = (const float*)d_in[7];  // [1]
    float* out = (float*)d_out;                 // [T]
    float* igw = (float*)d_ws;                  // igates workspace: T*384*4 = 96 MB

    // Phase 1: 4096*24 tiles, 1 wave/tile, 8 waves/block -> 12288 blocks.
    igates_gemm<<<12288, 256, 0, stream>>>(x, w_ih, b_ih, igw);
    // Phase 2: single workgroup, 24 waves, sequential over T.
    gru_scan<<<1, 768, 0, stream>>>(igw, xlin, h0, w_hh, b_n, lb, out);
}
